// NeuralShaderSegmentation_9500467658906
// MI455X (gfx1250) — matrix operand order repaired and checked by reading, not compiled, not run on hardware
//
#include <hip/hip_runtime.h>
#include <hip/hip_bf16.h>
#include <math.h>

typedef __attribute__((ext_vector_type(16))) _Float16 v16h;
typedef __attribute__((ext_vector_type(8)))  _Float16 h8;
typedef __attribute__((ext_vector_type(8)))  float    v8f;

#define RES   64
#define FDIM  32
#define NMOD  3
#define HID   64
#define NCLS  4
#define IN_PAD 160                 // IN_DIM=135 padded to 5*32
#define M_TOTAL (4*131072)         // 524288 points
#define TILES (M_TOTAL/16)         // 32768 (exact)
#define WAVES_PER_BLOCK 8
#define VOX (RES*RES*RES)          // 262144
#define NCH (NMOD*FDIM)            // 96

#define GRIDT_FLOATS ((size_t)VOX*NCH)        // 25,165,824 floats (~100.7 MB)
#define W0P_HALVES (4*5*512)
#define W1P_HALVES (4*2*512)
#define W2P_HALVES (1*2*512)

// ---- grid transpose: [mod][c][vox] -> [vox][c][mod], LDS-tiled, both sides coalesced ----
#define TVOX 128
__global__ __launch_bounds__(256)
void transpose_grids(const float* __restrict__ g, float* __restrict__ gt){
    __shared__ float tile[NCH][TVOX + 1];            // +1 pad: kill bank conflicts
    const size_t vox0 = (size_t)blockIdx.x * TVOX;
    const int tid = threadIdx.x;
    // read: src[mc][vox0+v], consecutive threads -> consecutive vox (coalesced)
    #pragma unroll
    for (int it = 0; it < NCH*TVOX/256; ++it){
        int idx = it*256 + tid;
        int mc = idx / TVOX, v = idx % TVOX;         // mc = mod*32 + c
        tile[mc][v] = g[(size_t)mc*VOX + vox0 + v];
    }
    __syncthreads();
    // write: dst[(vox*96) + c*3 + mod], consecutive threads -> consecutive q (coalesced)
    #pragma unroll
    for (int it = 0; it < NCH*TVOX/256; ++it){
        int idx = it*256 + tid;
        int v = idx / NCH, q = idx % NCH;            // q = c*3 + mod
        int c = q / NMOD, mod = q % NMOD;
        gt[(vox0 + v)*NCH + q] = tile[mod*FDIM + c][v];
    }
}

// -------- pack weights into WMMA B layout: lane -> N, halves -> K (split 0-15/16-31) --------
__global__ void pack_weights(const float* __restrict__ W0, const float* __restrict__ W1,
                             const float* __restrict__ W2,
                             _Float16* __restrict__ W0p, _Float16* __restrict__ W1p,
                             _Float16* __restrict__ W2p){
    int t = blockIdx.x * blockDim.x + threadIdx.x;
    if (t < W0P_HALVES){
        int g = t >> 9, idx = t & 511;
        int lane = idx >> 4, h = idx & 15;
        int kc = g % 5, nt = g / 5;
        int k = kc*32 + h + ((lane >= 16) ? 16 : 0);
        int n = nt*16 + (lane & 15);
        float v = (k < 135) ? W0[k*HID + n] : 0.f;
        W0p[t] = (_Float16)v;
    } else if (t < W0P_HALVES + W1P_HALVES){
        int u = t - W0P_HALVES;
        int g = u >> 9, idx = u & 511;
        int lane = idx >> 4, h = idx & 15;
        int kc = g & 1, nt = g >> 1;
        int k = kc*32 + h + ((lane >= 16) ? 16 : 0);
        int n = nt*16 + (lane & 15);
        W1p[u] = (_Float16)W1[k*HID + n];
    } else if (t < W0P_HALVES + W1P_HALVES + W2P_HALVES){
        int u = t - W0P_HALVES - W1P_HALVES;
        int g = u >> 9, idx = u & 511;
        int lane = idx >> 4, h = idx & 15;
        int k = g*32 + h + ((lane >= 16) ? 16 : 0);
        int n = lane & 15;
        float v = (n < (1+NCLS)) ? W2[k*(1+NCLS) + n] : 0.f;
        W2p[u] = (_Float16)v;
    }
}

// ---- A fragment from row-major f16 LDS tile: two aligned b128 LDS loads per lane ----
__device__ inline v16h load_A_lds(const _Float16* base, int lane, int stride_h, int kc){
    int row = lane & 15;
    int kb  = (lane & 16);                   // 0 or 16
    const _Float16* r = base + row*stride_h + kc*32 + kb;
    h8 lo = *(const h8*)(r);
    h8 hi = *(const h8*)(r + 8);
    return __builtin_shufflevector(lo, hi, 0,1,2,3,4,5,6,7,8,9,10,11,12,13,14,15);
}

__device__ inline v16h load_B_packed(const _Float16* __restrict__ p, int lane){
    const h8* q = (const h8*)(p + lane*16);
    h8 lo = q[0], hi = q[1];
    return __builtin_shufflevector(lo, hi, 0,1,2,3,4,5,6,7,8,9,10,11,12,13,14,15);
}

__device__ inline v8f wmma_f16(v16h a, v16h b, v8f c){
    return __builtin_amdgcn_wmma_f32_16x16x32_f16(false, a, false, b, (short)0, c, false, false);
}

// ---------------- fused sampling + PE + MLP: one wave per 16-point tile ----------------
__global__ __launch_bounds__(256)
void nerf_mlp(const float* __restrict__ pts, const float* __restrict__ gridT,
              const _Float16* __restrict__ W0p, const float* __restrict__ b0,
              const _Float16* __restrict__ W1p, const float* __restrict__ b1,
              const _Float16* __restrict__ W2p, const float* __restrict__ b2,
              float* __restrict__ out)
{
    __shared__ alignas(16) _Float16 sA[WAVES_PER_BLOCK][16*IN_PAD];  // 40 KB
    __shared__ alignas(16) _Float16 sB[WAVES_PER_BLOCK][16*HID];     // 16 KB
    const int lane = threadIdx.x & 31;
    const int wid  = threadIdx.x >> 5;
    const int tile = blockIdx.x * WAVES_PER_BLOCK + wid;  // TILES exact, EXEC stays full
    _Float16* hA = sA[wid];
    _Float16* hB = sB[wid];

    // ---- per-point coordinate math, lane-parallel (lane&15 owns point p) ----
    int gp0 = tile*16 + (lane & 15);
    float xl = pts[(size_t)gp0*3 + 0];
    float yl = pts[(size_t)gp0*3 + 1];
    float zl = pts[(size_t)gp0*3 + 2];
    float ixl = (xl + 1.f) * 0.5f * (RES - 1);
    float iyl = (yl + 1.f) * 0.5f * (RES - 1);
    float izl = (zl + 1.f) * 0.5f * (RES - 1);
    float fxfl = floorf(ixl), fyfl = floorf(iyl), fzfl = floorf(izl);
    float fxl = ixl - fxfl, fyl = iyl - fyfl, fzl = izl - fzfl;
    int   pkl = (int)fxfl | ((int)fyfl << 8) | ((int)fzfl << 16);

    // ---- Phase 1: trilinear features (lane = channel) + positional encoding ----
    for (int p = 0; p < 16; ++p){
        float x  = __shfl(xl, p),  y  = __shfl(yl, p),  z  = __shfl(zl, p);
        float fx = __shfl(fxl, p), fy = __shfl(fyl, p), fz = __shfl(fzl, p);
        int   pk = __shfl(pkl, p);
        int ix0 = pk & 0xFF, iy0 = (pk >> 8) & 0xFF, iz0 = (pk >> 16) & 0xFF;
        float wx[2] = {1.f - fx, fx};
        float wy[2] = {1.f - fy, fy};
        float wz[2] = {1.f - fz, fz};
        float a0 = 0.f, a1 = 0.f, a2 = 0.f;
        #pragma unroll
        for (int corner = 0; corner < 8; ++corner){
            int dx = corner & 1, dy = (corner >> 1) & 1, dz = (corner >> 2) & 1;
            int xc = ix0 + dx, yc = iy0 + dy, zc = iz0 + dz;
            float w = wx[dx] * wy[dy] * wz[dz];
            bool valid = ((unsigned)xc < RES) && ((unsigned)yc < RES) && ((unsigned)zc < RES);
            if (valid){  // uniform per wave; one contiguous 384B b96 wave-load per corner
                size_t vox = ((size_t)zc*RES + yc)*RES + xc;
                const float* gb = gridT + (vox*FDIM + lane)*NMOD;
                float g0 = gb[0], g1 = gb[1], g2 = gb[2];
                a0 += w * g0;  a1 += w * g1;  a2 += w * g2;
            }
        }
        hA[p*IN_PAD + 0*FDIM + lane] = (_Float16)a0;
        hA[p*IN_PAD + 1*FDIM + lane] = (_Float16)a1;
        hA[p*IN_PAD + 2*FDIM + lane] = (_Float16)a2;
        #pragma unroll
        for (int jj = 0; jj < 2; ++jj){       // cols 96..159: 39 PE values + zero pad
            int j = lane + jj*32;
            float v = 0.f;
            if (j < 39){
                if (j < 3) v = (j == 0) ? x : ((j == 1) ? y : z);
                else {
                    int f = (j - 3) / 6, r = (j - 3) % 6, d = r % 3;
                    float pv = (d == 0) ? x : ((d == 1) ? y : z);
                    float a = 3.14159265358979323846f * (float)(1 << f) * pv;
                    v = (r < 3) ? __sinf(a) : __cosf(a);   // native v_sin/v_cos
                }
            }
            hA[p*IN_PAD + 96 + j] = (_Float16)v;
        }
    }
    __syncthreads();

    // ---- Layer 0: [16x160] @ [160x64], 5 K-chunks x 4 N-tiles ----
    v8f c0[4];
    #pragma unroll
    for (int nt = 0; nt < 4; ++nt) c0[nt] = (v8f){};
    #pragma unroll
    for (int kc = 0; kc < 5; ++kc){
        v16h a = load_A_lds(hA, lane, IN_PAD, kc);
        #pragma unroll
        for (int nt = 0; nt < 4; ++nt){
            v16h b = load_B_packed(W0p + (size_t)(nt*5 + kc)*512, lane);
            c0[nt] = wmma_f16(a, b, c0[nt]);
        }
    }
    {   // bias + ReLU, C-layout -> row-major LDS
        int col = lane & 15;
        int mh  = (lane & 16) >> 1;
        #pragma unroll
        for (int nt = 0; nt < 4; ++nt){
            int n = nt*16 + col;
            float bb = b0[n];
            #pragma unroll
            for (int v = 0; v < 8; ++v)
                hB[(v + mh)*HID + n] = (_Float16)fmaxf(c0[nt][v] + bb, 0.f);
        }
    }
    __syncthreads();

    // ---- Layer 1: [16x64] @ [64x64], 2 K-chunks x 4 N-tiles ----
    v8f c1[4];
    #pragma unroll
    for (int nt = 0; nt < 4; ++nt) c1[nt] = (v8f){};
    #pragma unroll
    for (int kc = 0; kc < 2; ++kc){
        v16h a = load_A_lds(hB, lane, HID, kc);
        #pragma unroll
        for (int nt = 0; nt < 4; ++nt){
            v16h b = load_B_packed(W1p + (size_t)(nt*2 + kc)*512, lane);
            c1[nt] = wmma_f16(a, b, c1[nt]);
        }
    }
    __syncthreads();
    {
        int col = lane & 15;
        int mh  = (lane & 16) >> 1;
        #pragma unroll
        for (int nt = 0; nt < 4; ++nt){
            int n = nt*16 + col;
            float bb = b1[n];
            #pragma unroll
            for (int v = 0; v < 8; ++v)
                hB[(v + mh)*HID + n] = (_Float16)fmaxf(c1[nt][v] + bb, 0.f);
        }
    }
    __syncthreads();

    // ---- Layer 2: [16x64] @ [64x16(5 valid)] ----
    v8f c2 = (v8f){};
    #pragma unroll
    for (int kc = 0; kc < 2; ++kc){
        v16h a = load_A_lds(hB, lane, HID, kc);
        v16h b = load_B_packed(W2p + (size_t)kc*512, lane);
        c2 = wmma_f16(a, b, c2);
    }

    // ---- Output: density = softplus(col 0), seg logits = cols 1..4 ----
    {
        int n  = lane & 15;
        int mh = (lane & 16) >> 1;
        if (n < 1 + NCLS){
            float bb = b2[n];
            #pragma unroll
            for (int v = 0; v < 8; ++v){
                float xv = c2[v] + bb;
                size_t gm = (size_t)tile*16 + v + mh;
                if (n == 0){
                    float sp = (xv > 0.f) ? (xv + log1pf(expf(-xv))) : log1pf(expf(xv));
                    out[gm] = sp;
                } else {
                    out[(size_t)M_TOTAL + gm*NCLS + (n - 1)] = xv;
                }
            }
        }
    }
}

extern "C" void kernel_launch(void* const* d_in, const int* in_sizes, int n_in,
                              void* d_out, int out_size, void* d_ws, size_t ws_size,
                              hipStream_t stream) {
    const float* pts   = (const float*)d_in[0];
    const float* grids = (const float*)d_in[1];
    const float* W0    = (const float*)d_in[2];
    const float* b0    = (const float*)d_in[3];
    const float* W1    = (const float*)d_in[4];
    const float* b1    = (const float*)d_in[5];
    const float* W2    = (const float*)d_in[6];
    const float* b2    = (const float*)d_in[7];
    float* out = (float*)d_out;

    float*    gridT = (float*)d_ws;
    _Float16* W0p   = (_Float16*)(gridT + GRIDT_FLOATS);
    _Float16* W1p   = W0p + W0P_HALVES;
    _Float16* W2p   = W1p + W1P_HALVES;

    transpose_grids<<<VOX/TVOX, 256, 0, stream>>>(grids, gridT);
    {
        int total = W0P_HALVES + W1P_HALVES + W2P_HALVES;
        pack_weights<<<(total + 255) / 256, 256, 0, stream>>>(W0, W1, W2, W0p, W1p, W2p);
    }
    nerf_mlp<<<TILES / WAVES_PER_BLOCK, 256, 0, stream>>>(pts, gridT, W0p, b0, W1p, b1, W2p, b2, out);
}
